// RewardModulatedSTDPLearner_42039139893632
// MI455X (gfx1250) — matrix operand order repaired and checked by reading, not compiled, not run on hardware
//
#include <hip/hip_runtime.h>
#include <hip/hip_bf16.h>
#include <stdint.h>

typedef __attribute__((ext_vector_type(16))) __bf16 v16bf;
typedef __attribute__((ext_vector_type(8)))  __bf16 v8bf;
typedef __attribute__((ext_vector_type(8)))  float  v8f;

#define B_   8
#define T_   2048
#define N_   1024            // PRE_N == POST_N
#define K_   (B_ * T_)       // 16384 contraction length
#define KSPLIT_ 8
#define KCH_ (K_ / KSPLIT_)  // 2048 per K-chunk
#define ITERS_ (KCH_ / 32)   // 64 K-steps per block

// LDS tile geometry: per K-step stage 4 arrays x 128 rows x 32 K (bf16).
// Row pitch padded to 40 elements (80 B) => bank stride 20, conflict-free
// for 16-lane row-strided ds_load_b128 fragment reads.
#define PITCH_ 40                    // elements per row
#define ARRE_  (128 * PITCH_)        // 5120 elements per array
#define BUFE_  (4 * ARRE_)           // 20480 elements per buffer
#define BUFBYTES_ (BUFE_ * 2)        // 40960 bytes
#define SMEM_BYTES_ (2 * BUFBYTES_)  // 81920 bytes (double buffered)

// Workspace layout: four bf16 matrices, each [N_][K_] (K contiguous):
//   slot 0: pre_spikes^T   (unscaled)      -> B operand of GEMM2
//   slot 1: pre_trace^T    (* +r*0.01/8)   -> B operand of GEMM1
//   slot 2: post_spikes^T  (unscaled)      -> A operand of GEMM1
//   slot 3: post_trace^T   (* -r*0.01/8)   -> A operand of GEMM2

// ---------------------------------------------------------------------------
// Stage 1: exponential traces + bf16 convert + transpose to [N, K] layout.
// ---------------------------------------------------------------------------
__global__ __launch_bounds__(256) void stdp_trace_transpose(
    const float* __restrict__ pre, const float* __restrict__ post,
    const float* __restrict__ reward, __bf16* __restrict__ ws)
{
  const int tid  = blockIdx.x * blockDim.x + threadIdx.x;  // 0 .. 2*B*N-1
  const int half = tid / (B_ * N_);                        // 0=pre, 1=post
  const int rem  = tid % (B_ * N_);
  const int b = rem / N_;
  const int n = rem % N_;

  const float r     = reward[0];
  const float decay = 0.95122942450071400910f;             // exp(-1/20)
  const float scale = half ? (-0.01f / 8.0f) * r : (0.01f / 8.0f) * r;

  const float* __restrict__ src = (half ? post : pre) + (size_t)b * T_ * N_ + n;
  __bf16* __restrict__ spkT = ws + (size_t)(half ? 2 : 0) * N_ * K_
                                 + (size_t)n * K_ + (size_t)b * T_;
  __bf16* __restrict__ trT  = ws + (size_t)(half ? 3 : 1) * N_ * K_
                                 + (size_t)n * K_ + (size_t)b * T_;

  float tr = 0.0f;
  for (int t0 = 0; t0 < T_; t0 += 8) {
    v8bf sv, tv;
#pragma unroll
    for (int u = 0; u < 8; ++u) {
      const int t = t0 + u;
      const float s = src[(size_t)t * N_];
      if (t > 0) tr = decay * tr + s;       // trace[0] == 0, skips spikes[0]
      sv[u] = (__bf16)s;
      tv[u] = (__bf16)(scale * tr);
    }
    *(v8bf*)(spkT + t0) = sv;
    *(v8bf*)(trT  + t0) = tv;
  }
}

// ---------------------------------------------------------------------------
// Stage 2: fused dual GEMM. Block = 4 waves = 128x128 output tile (2x2
// quadrants of 64x64). Per K-step of 32: async-copy 32 KB of operands into
// LDS (double buffered, ASYNCcnt), read WMMA fragments with ds_load_b128,
// issue 32 v_wmma_f32_16x16x32_bf16 per wave. Split-K x8, f32 atomics.
// ---------------------------------------------------------------------------
__global__ __launch_bounds__(128) void stdp_wmma_gemm(
    const __bf16* __restrict__ ws, float* __restrict__ out)
{
  extern __shared__ __bf16 smem[];   // SMEM_BYTES_ = 80 KB dynamic LDS

  const int tidb = threadIdx.x;
  const int lane = tidb & 31;
  const int wave = tidb >> 5;
  const int blk  = blockIdx.x;                 // 0 .. 511
  const int tile = blk & 63;                   // 8x8 grid of 128x128 tiles
  const int kc   = blk >> 6;                   // K chunk 0..7
  const int tm   = (tile >> 3) << 7;
  const int tn   = (tile & 7) << 7;
  const int wm   = (wave >> 1) << 6;           // wave quadrant in tile
  const int wn   = (wave & 1) << 6;
  const int l15  = lane & 15;
  const int hi   = lane >> 4;
  const int k0   = kc * KCH_;

  // arr order in LDS: 0=postT(A1), 1=preTr(B1), 2=postTr(A2), 3=preT(B2)
  const __bf16* gbase[4] = { ws + 2ull * N_ * K_, ws + 1ull * N_ * K_,
                             ws + 3ull * N_ * K_, ws + 0ull * N_ * K_ };

  // Per-thread staging assignment: 4 threads per row (16B chunks), 32 row
  // groups x 4 passes x 4 arrays = 16 async b128 issues per thread per step.
  const int rrow  = tidb >> 2;                 // 0..31
  const int chunk = tidb & 3;                  // 16B chunk within 64B row
  uint64_t ga[16];
  uint32_t la[16];
#pragma unroll
  for (int arr = 0; arr < 4; ++arr)
#pragma unroll
    for (int rr = 0; rr < 4; ++rr) {
      const int row  = rr * 32 + rrow;
      const int grow = ((arr & 1) ? tn : tm) + row;   // arr 0,2 -> A rows; 1,3 -> B rows
      ga[arr * 4 + rr] = (uint64_t)(uintptr_t)(gbase[arr] + (size_t)grow * K_ + k0)
                         + (uint64_t)chunk * 16;
      la[arr * 4 + rr] = (uint32_t)(uintptr_t)&smem[arr * ARRE_ + row * PITCH_]
                         + (uint32_t)chunk * 16;
    }

  auto stage = [&](uint32_t bufByte) {
#pragma unroll
    for (int q = 0; q < 16; ++q) {
      asm volatile("global_load_async_to_lds_b128 %0, %1, off"
                   :: "v"(la[q] + bufByte), "v"(ga[q]) : "memory");
      ga[q] += 64;   // advance 32 bf16 in K
    }
  };

  v8f c[4][4];
#pragma unroll
  for (int i = 0; i < 4; ++i)
#pragma unroll
    for (int j = 0; j < 4; ++j)
#pragma unroll
      for (int e = 0; e < 8; ++e) c[i][j][e] = 0.0f;

  // Prologue: fill buffer 0.
  stage(0u);
  asm volatile("s_wait_asynccnt 0x0" ::: "memory");
  __syncthreads();

  int cur = 0;
  for (int it = 0; it < ITERS_; ++it) {
    if (it + 1 < ITERS_) stage((cur ^ 1) ? BUFBYTES_ : 0u);   // overlap copy

    const __bf16* sb = smem + (uint32_t)cur * BUFE_;
    v16bf a[4], bfrag[4];

    // ---- GEMM1: post_spikes (arr0) x pre_trace (arr1) ----
#pragma unroll
    for (int i = 0; i < 4; ++i) {
      const __bf16* pa = sb + 0 * ARRE_ + (uint32_t)(wm + i * 16 + l15) * PITCH_ + hi * 8;
      v8bf alo = *(const v8bf*)pa;
      v8bf ahi = *(const v8bf*)(pa + 16);
      a[i] = __builtin_shufflevector(alo, ahi, 0,1,2,3,4,5,6,7,8,9,10,11,12,13,14,15);
      const __bf16* pb = sb + 1 * ARRE_ + (uint32_t)(wn + i * 16 + l15) * PITCH_ + hi * 8;
      v8bf blo = *(const v8bf*)pb;
      v8bf bhi = *(const v8bf*)(pb + 16);
      bfrag[i] = __builtin_shufflevector(blo, bhi, 0,1,2,3,4,5,6,7,8,9,10,11,12,13,14,15);
    }
#pragma unroll
    for (int i = 0; i < 4; ++i)
#pragma unroll
      for (int j = 0; j < 4; ++j)
        c[i][j] = __builtin_amdgcn_wmma_f32_16x16x32_bf16(
            false, a[i], false, bfrag[j], (short)0, c[i][j], false, false);

    // ---- GEMM2: post_trace (arr2) x pre_spikes (arr3) ----
#pragma unroll
    for (int i = 0; i < 4; ++i) {
      const __bf16* pa = sb + 2 * ARRE_ + (uint32_t)(wm + i * 16 + l15) * PITCH_ + hi * 8;
      v8bf alo = *(const v8bf*)pa;
      v8bf ahi = *(const v8bf*)(pa + 16);
      a[i] = __builtin_shufflevector(alo, ahi, 0,1,2,3,4,5,6,7,8,9,10,11,12,13,14,15);
      const __bf16* pb = sb + 3 * ARRE_ + (uint32_t)(wn + i * 16 + l15) * PITCH_ + hi * 8;
      v8bf blo = *(const v8bf*)pb;
      v8bf bhi = *(const v8bf*)(pb + 16);
      bfrag[i] = __builtin_shufflevector(blo, bhi, 0,1,2,3,4,5,6,7,8,9,10,11,12,13,14,15);
    }
#pragma unroll
    for (int i = 0; i < 4; ++i)
#pragma unroll
      for (int j = 0; j < 4; ++j)
        c[i][j] = __builtin_amdgcn_wmma_f32_16x16x32_bf16(
            false, a[i], false, bfrag[j], (short)0, c[i][j], false, false);

    asm volatile("s_wait_asynccnt 0x0" ::: "memory");  // next buffer landed
    __syncthreads();
    cur ^= 1;
  }

  // C/D layout: VGPR e holds M = e + 8*hi, N = l15 (per 16x16 tile).
#pragma unroll
  for (int i = 0; i < 4; ++i)
#pragma unroll
    for (int j = 0; j < 4; ++j)
#pragma unroll
      for (int e = 0; e < 8; ++e) {
        const int m = tm + wm + i * 16 + hi * 8 + e;
        const int q = tn + wn + j * 16 + l15;
        __hip_atomic_fetch_add(&out[(size_t)m * N_ + q], c[i][j][e],
                               __ATOMIC_RELAXED, __HIP_MEMORY_SCOPE_AGENT);
      }
}

// ---------------------------------------------------------------------------
extern "C" void kernel_launch(void* const* d_in, const int* in_sizes, int n_in,
                              void* d_out, int out_size, void* d_ws, size_t ws_size,
                              hipStream_t stream) {
  (void)in_sizes; (void)n_in; (void)ws_size;
  // d_in[0] = weights (unused by the reference output)
  const float* pre    = (const float*)d_in[1];
  const float* post   = (const float*)d_in[2];
  const float* reward = (const float*)d_in[3];
  float* out = (float*)d_out;
  __bf16* ws = (__bf16*)d_ws;   // needs 128 MB

  hipMemsetAsync(d_out, 0, (size_t)out_size * sizeof(float), stream);
  stdp_trace_transpose<<<dim3((2 * B_ * N_) / 256), dim3(256), 0, stream>>>(
      pre, post, reward, ws);
  stdp_wmma_gemm<<<dim3(64 * KSPLIT_), dim3(128), SMEM_BYTES_, stream>>>(ws, out);
}
